// TrafficTransformerDecoder_16338055594427
// MI455X (gfx1250) — matrix operand order repaired and checked by reading, not compiled, not run on hardware
//
#include <hip/hip_runtime.h>

typedef unsigned short u16;
typedef __attribute__((ext_vector_type(16))) __bf16 v16bf;
typedef __attribute__((ext_vector_type(8)))  float  v8f;

// ---------- helpers ----------
__device__ __forceinline__ u16 f2bf(float f) {
  union { float f; unsigned u; } v; v.f = f;
  unsigned u = v.u;
  unsigned r = u + 0x7FFFu + ((u >> 16) & 1u);   // RNE
  return (u16)(r >> 16);
}

// A fragment: 16x32 bf16. lanes 0-15: row M=lane, K = kb+[0..7] and kb+[16..23];
// lanes 16-31: row M=lane-16, K = kb+[8..15] and kb+[24..31]. (ISA 7.12.2)
__device__ __forceinline__ v16bf load_a_frag(const u16* s, int ld, int m0, int kb, int lane) {
  int row = m0 + (lane & 15);
  int ko  = kb + ((lane < 16) ? 0 : 8);
  const u16* p = s + row * ld + ko;
  v16bf a;
  ((uint4*)&a)[0] = *(const uint4*)(p);
  ((uint4*)&a)[1] = *(const uint4*)(p + 16);
  return a;
}
// B fragment from LDS holding B^T (s[n][k]): lanes 0-15: col=lane, K=kb+[0..7] and kb+[16..23];
// lanes 16-31: col=lane-16, K=kb+[8..15] and kb+[24..31].
__device__ __forceinline__ v16bf load_b_frag(const u16* s, int ld, int n0, int kb, int lane) {
  int col = n0 + (lane & 15);
  int ko  = kb + ((lane < 16) ? 0 : 8);
  const u16* p = s + col * ld + ko;
  v16bf b;
  ((uint4*)&b)[0] = *(const uint4*)(p);
  ((uint4*)&b)[1] = *(const uint4*)(p + 16);
  return b;
}
__device__ __forceinline__ v8f wmma_bf16(v16bf a, v16bf b, v8f c) {
  return __builtin_amdgcn_wmma_f32_16x16x32_bf16(false, a, false, b, (short)0, c, false, false);
}

// ---------- elementwise kernels ----------
__global__ void cvt_f32_bf16(const float* __restrict__ in, u16* __restrict__ out, long long n) {
  long long i = (long long)blockIdx.x * blockDim.x + threadIdx.x;
  if (i < n) out[i] = f2bf(in[i]);
}

// transpose-convert: in (R x C, f32, row-major) -> out (C x R, bf16). blockIdx.z batches.
__global__ void cvtT_f32_bf16(const float* __restrict__ in, u16* __restrict__ out, int R, int C) {
  long long per = (long long)R * C;
  long long i = (long long)blockIdx.x * blockDim.x + threadIdx.x;
  if (i >= per) return;
  const float* ib = in + (long long)blockIdx.z * per;
  u16* ob = out + (long long)blockIdx.z * per;
  int r = (int)(i / C), c = (int)(i % C);
  ob[(long long)c * R + r] = f2bf(ib[i]);
}

// out[r,c] = bf16(x[r,c] + pos[((r/posDiv)*posMod + r%posMod), c]),  width N
__global__ void addpos_bf16(const float* __restrict__ x, const float* __restrict__ pos,
                            u16* __restrict__ out, long long total, int N, int posDiv, int posMod) {
  long long i = (long long)blockIdx.x * blockDim.x + threadIdx.x;
  if (i >= total) return;
  int col = (int)(i % N);
  long long row = i / N;
  long long pr = (row / posDiv) * posMod + (row % posMod);
  out[i] = f2bf(x[i] + pos[pr * N + col]);
}

// ---------- IDW adjacency: one block per batch, 128 threads = rows ----------
__global__ __launch_bounds__(128) void adj_kernel(const float* __restrict__ locs, u16* __restrict__ adjb) {
  __shared__ float lx[128], ly[128];
  int b = blockIdx.x, m = threadIdx.x;
  lx[m] = locs[(b * 128 + m) * 2 + 0];
  ly[m] = locs[(b * 128 + m) * 2 + 1];
  __syncthreads();
  float xm = lx[m], ym = ly[m], s = 0.f;
  for (int n = 0; n < 128; n++) {
    if (n == m) continue;
    float dx = xm - lx[n], dy = ym - ly[n];
    s += 1.f / sqrtf(dx * dx + dy * dy + 1e-8f);
  }
  float inv = 1.f / s;
  u16* row = adjb + ((long long)(b * 128 + m)) * 128;
  for (int n = 0; n < 128; n++) {
    float w = 0.f;
    if (n != m) {
      float dx = xm - lx[n], dy = ym - ly[n];
      w = inv / sqrtf(dx * dx + dy * dy + 1e-8f);
    }
    row[n] = f2bf(w);
  }
}

// ---------- generic bf16 WMMA GEMM: C = epi(A @ Bt^T) ----------
// A: M x K bf16 row-major. Bt: N x K bf16 row-major (pre-transposed B).
// Double-buffered LDS, register prefetch of next K tile.
// Epilogue: +bias, relu, +pos (broadcast), +res(f32), write Cf(f32) and/or Cb(bf16).
#define MBLK 128
#define NBLK 128
#define KSTEP 32
#define KPAD 40
__global__ __launch_bounds__(256) void gemm_bf16(
    const u16* __restrict__ A, const u16* __restrict__ Bt,
    const float* __restrict__ bias, const float* __restrict__ res,
    const float* __restrict__ pos, int posDiv, int posMod,
    float* __restrict__ Cf, u16* __restrict__ Cb,
    int M, int N, int K, int relu,
    long long strideA, long long strideB, long long strideC)
{
  __shared__ u16 sA[2][MBLK][KPAD];
  __shared__ u16 sB[2][NBLK][KPAD];
  int bz = blockIdx.z;
  const u16* Ab  = A  + (long long)bz * strideA;
  const u16* Btb = Bt + (long long)bz * strideB;
  long long cbase = (long long)bz * strideC;
  int m0 = blockIdx.y * MBLK;
  int n0 = blockIdx.x * NBLK;
  int tid = threadIdx.x;
  int lane = tid & 31;
  int wave = tid >> 5;
  int wm = (wave >> 1) * 32;   // 0,32,64,96
  int wn = (wave & 1) * 64;    // 0,64
  v8f acc[2][4] = {};

  // staging: 2 threads per row, 16 u16 each; covers 128x32 for A and B
  int r = tid >> 1;
  int c = (tid & 1) * 16;
  const u16* gA = Ab  + (long long)(m0 + r) * K + c;
  const u16* gB = Btb + (long long)(n0 + r) * K + c;
  uint4 ra0 = *(const uint4*)(gA);
  uint4 ra1 = *(const uint4*)(gA + 8);
  uint4 rb0 = *(const uint4*)(gB);
  uint4 rb1 = *(const uint4*)(gB + 8);

  int cur = 0;
  for (int k0 = 0; k0 < K; k0 += KSTEP) {
    *(uint4*)&sA[cur][r][c]     = ra0;
    *(uint4*)&sA[cur][r][c + 8] = ra1;
    *(uint4*)&sB[cur][r][c]     = rb0;
    *(uint4*)&sB[cur][r][c + 8] = rb1;
    __syncthreads();
    if (k0 + KSTEP < K) {
      const u16* nA = gA + k0 + KSTEP;
      const u16* nB = gB + k0 + KSTEP;
      ra0 = *(const uint4*)(nA); ra1 = *(const uint4*)(nA + 8);
      rb0 = *(const uint4*)(nB); rb1 = *(const uint4*)(nB + 8);
    }
    if (k0 + 2 * KSTEP < K) {
      __builtin_prefetch(gA + k0 + 2 * KSTEP, 0, 1);
      __builtin_prefetch(gB + k0 + 2 * KSTEP, 0, 1);
    }
    v16bf af0 = load_a_frag(&sA[cur][0][0], KPAD, wm,      0, lane);
    v16bf af1 = load_a_frag(&sA[cur][0][0], KPAD, wm + 16, 0, lane);
    v16bf bf0 = load_b_frag(&sB[cur][0][0], KPAD, wn,      0, lane);
    v16bf bf1 = load_b_frag(&sB[cur][0][0], KPAD, wn + 16, 0, lane);
    v16bf bf2 = load_b_frag(&sB[cur][0][0], KPAD, wn + 32, 0, lane);
    v16bf bf3 = load_b_frag(&sB[cur][0][0], KPAD, wn + 48, 0, lane);
    acc[0][0] = wmma_bf16(af0, bf0, acc[0][0]);
    acc[0][1] = wmma_bf16(af0, bf1, acc[0][1]);
    acc[0][2] = wmma_bf16(af0, bf2, acc[0][2]);
    acc[0][3] = wmma_bf16(af0, bf3, acc[0][3]);
    acc[1][0] = wmma_bf16(af1, bf0, acc[1][0]);
    acc[1][1] = wmma_bf16(af1, bf1, acc[1][1]);
    acc[1][2] = wmma_bf16(af1, bf2, acc[1][2]);
    acc[1][3] = wmma_bf16(af1, bf3, acc[1][3]);
    cur ^= 1;
  }

  int rh = (lane < 16) ? 0 : 8;
  int cn = lane & 15;
  for (int i = 0; i < 2; i++)
    for (int j = 0; j < 4; j++) {
      int colg = n0 + wn + j * 16 + cn;
      float bv = bias ? bias[colg] : 0.f;
      for (int e = 0; e < 8; e++) {
        int rowg = m0 + wm + i * 16 + rh + e;
        float v = acc[i][j][e] + bv;
        if (relu) v = fmaxf(v, 0.f);
        if (pos) {
          int pr = (rowg / posDiv) * posMod + (rowg % posMod);
          v += pos[(long long)pr * N + colg];
        }
        long long idx = cbase + (long long)rowg * N + colg;
        if (res) v += res[idx];
        if (Cf) Cf[idx] = v;
        if (Cb) Cb[idx] = f2bf(v);
      }
    }
}

// ---------- attention: one wave per (sequence, head). T=48, hd=32, D=256 ----------
__global__ __launch_bounds__(32) void attn_kernel(
    const u16* __restrict__ Q, const u16* __restrict__ Km,
    const u16* __restrict__ Vm, u16* __restrict__ O, int causal)
{
  __shared__ u16  sQ[48][40];
  __shared__ u16  sK[48][40];
  __shared__ u16  sVt[32][72];   // V^T, K padded 48->64 (+8 pad)
  __shared__ float sS[48][48];
  __shared__ u16  sP[48][72];
  int seq = blockIdx.x, head = blockIdx.y;
  int lane = threadIdx.x;
  long long base = ((long long)seq * 48) * 256 + head * 32;

  for (int i = lane; i < 48 * 4; i += 32) {       // Q,K: 48x32, 8 u16 chunks
    int r = i >> 2, c = (i & 3) * 8;
    *(uint4*)&sQ[r][c] = *(const uint4*)(Q  + base + r * 256 + c);
    *(uint4*)&sK[r][c] = *(const uint4*)(Km + base + r * 256 + c);
  }
  for (int i = lane; i < 48 * 32; i += 32) {      // V^T
    int k = i >> 5, d = i & 31;
    sVt[d][k] = Vm[base + k * 256 + d];
  }
  for (int i = lane; i < 32 * 16; i += 32) {      // zero K pad 48..63
    int d = i >> 4;
    sVt[d][48 + (i & 15)] = 0;
  }
  for (int i = lane; i < 48 * 16; i += 32) {      // zero P cols 48..63
    int r = i >> 4;
    sP[r][48 + (i & 15)] = 0;
  }
  __syncthreads();

  const float scale = 0.17677669529663687f;       // 1/sqrt(32)
  int rh = (lane < 16) ? 0 : 8;
  int cn = lane & 15;
  for (int mi = 0; mi < 3; mi++) {
    v16bf a = load_a_frag(&sQ[0][0], 40, mi * 16, 0, lane);
    for (int ni = 0; ni < 3; ni++) {
      v16bf b = load_b_frag(&sK[0][0], 40, ni * 16, 0, lane);
      v8f c = {};
      c = wmma_bf16(a, b, c);
      for (int e = 0; e < 8; e++)
        sS[mi * 16 + rh + e][ni * 16 + cn] = c[e] * scale;
    }
  }
  __syncthreads();

  for (int r = lane; r < 48; r += 32) {
    int lim = causal ? (r + 1) : 48;
    float mx = -1e30f;
    for (int n = 0; n < lim; n++) mx = fmaxf(mx, sS[r][n]);
    float sum = 0.f;
    for (int n = 0; n < lim; n++) { float e = __expf(sS[r][n] - mx); sS[r][n] = e; sum += e; }
    float inv = 1.f / sum;
    for (int n = 0; n < 48; n++) sP[r][n] = (n < lim) ? f2bf(sS[r][n] * inv) : (u16)0;
  }
  __syncthreads();

  for (int mi = 0; mi < 3; mi++) {
    for (int nj = 0; nj < 2; nj++) {
      v8f c = {};
      for (int ks = 0; ks < 2; ks++) {
        v16bf a = load_a_frag(&sP[0][0],  72, mi * 16, ks * 32, lane);
        v16bf b = load_b_frag(&sVt[0][0], 72, nj * 16, ks * 32, lane);
        c = wmma_bf16(a, b, c);
      }
      for (int e = 0; e < 8; e++)
        O[base + (long long)(mi * 16 + rh + e) * 256 + nj * 16 + cn] = f2bf(c[e]);
    }
  }
}

// ---------- LayerNorm over D=256: 8 waves/block, one wave per row ----------
__global__ __launch_bounds__(256) void ln_kernel(
    const float* __restrict__ x, const float* __restrict__ g, const float* __restrict__ b,
    float* __restrict__ outf, u16* __restrict__ outb)
{
  int row = blockIdx.x * 8 + (threadIdx.x >> 5);
  int lane = threadIdx.x & 31;
  const float* xr = x + (long long)row * 256;
  float v[8]; float s = 0.f;
  for (int e = 0; e < 8; e++) { v[e] = xr[lane + e * 32]; s += v[e]; }
  for (int o = 16; o > 0; o >>= 1) s += __shfl_xor(s, o, 32);
  float mean = s * (1.f / 256.f);
  float q = 0.f;
  for (int e = 0; e < 8; e++) { float d = v[e] - mean; q += d * d; }
  for (int o = 16; o > 0; o >>= 1) q += __shfl_xor(q, o, 32);
  float inv = rsqrtf(q * (1.f / 256.f) + 1e-5f);
  for (int e = 0; e < 8; e++) {
    int cc = lane + e * 32;
    float o = (v[e] - mean) * inv * g[cc] + b[cc];
    long long idx = (long long)row * 256 + cc;
    if (outf) outf[idx] = o;
    if (outb) outb[idx] = f2bf(o);
  }
}

// ---------- launcher ----------
extern "C" void kernel_launch(void* const* d_in, const int* in_sizes, int n_in,
                              void* d_out, int out_size, void* d_ws, size_t ws_size,
                              hipStream_t stream) {
  (void)in_sizes; (void)n_in; (void)out_size; (void)ws_size;
  const float* dec_in  = (const float*)d_in[0];
  const float* enc_out = (const float*)d_in[1];
  const float* locs    = (const float*)d_in[2];
  const float* enc_pos = (const float*)d_in[3];
  const float* dec_pos = (const float*)d_in[4];
  const float* gcn_W = (const float*)d_in[5];
  const float* gcn_b = (const float*)d_in[6];
  const float* fc1_W = (const float*)d_in[7];
  const float* fc1_b = (const float*)d_in[8];
  const float* W[8]; const float* bb[8];       // sa_{q,k,v,o}, ca_{q,k,v,o}
  for (int i = 0; i < 8; i++) { W[i] = (const float*)d_in[9 + 2*i]; bb[i] = (const float*)d_in[10 + 2*i]; }
  const float* ff_W1 = (const float*)d_in[25];
  const float* ff_b1 = (const float*)d_in[26];
  const float* ff_W2 = (const float*)d_in[27];
  const float* ff_b2 = (const float*)d_in[28];
  const float* ln_g[3] = { (const float*)d_in[29], (const float*)d_in[31], (const float*)d_in[33] };
  const float* ln_b[3] = { (const float*)d_in[30], (const float*)d_in[32], (const float*)d_in[34] };

  const int B = 32, NS = 128, T = 48, NF = 32, D = 256, DFF = 1024;
  const long long RT = (long long)B * NS * T;          // 196608 rows
  const long long XE = RT * D;                         // 50,331,648 elems

  char* ws = (char*)d_ws;
  size_t off = 0;
  auto alloc = [&](size_t bytes) -> char* {
    char* p = ws + off; off += (bytes + 255) & ~(size_t)255; return p;
  };
  // all weight matrices stored TRANSPOSED (N x K) in bf16
  u16* wGCN = (u16*)alloc((size_t)NF * D * 2);          // 256 x 32
  u16* wFC1 = (u16*)alloc((size_t)D * D * 2);           // 256 x 256
  u16* wAT[8];
  for (int i = 0; i < 8; i++) wAT[i] = (u16*)alloc((size_t)D * D * 2);
  u16* wFF1 = (u16*)alloc((size_t)D * DFF * 2);         // 1024 x 256
  u16* wFF2 = (u16*)alloc((size_t)DFF * D * 2);         // 256 x 1024
  float* x32 = (float*)alloc((size_t)XE * 4);
  float* y32 = (float*)alloc((size_t)XE * 4);
  u16*  xb  = (u16*)alloc((size_t)XE * 2);
  u16*  big = (u16*)alloc((size_t)4 * XE * 2);
  u16* qb = big;
  u16* kb = big + XE;
  u16* vb = big + 2 * XE;
  u16* mb = big + 3 * XE;
  u16* h1 = big;                                        // FF hidden overlays q/k/v/m
  // phase-A temporaries inside y32 region
  u16* adjb  = (u16*)y32;                               // 32*128*128
  u16* decTb = adjb + (size_t)B * NS * NS;              // 32 x (1536 x 128), transposed per batch
  u16* aggb  = decTb + (size_t)B * NS * T * NF;         // 32*128*1536

  auto cvt = [&](const float* src, u16* dst, long long n) {
    cvt_f32_bf16<<<dim3((unsigned)((n + 255) / 256)), dim3(256), 0, stream>>>(src, dst, n);
  };
  auto cvtT = [&](const float* src, u16* dst, int R, int C, int batch) {
    cvtT_f32_bf16<<<dim3((unsigned)(((long long)R * C + 255) / 256), 1, batch),
                    dim3(256), 0, stream>>>(src, dst, R, C);
  };
  auto gemm = [&](const u16* A, const u16* Bt, const float* bias, const float* res,
                  const float* pos, int pD, int pM, float* Cf, u16* Cb,
                  int M, int N, int K, int relu, long long sA, long long sB, long long sC, int batch) {
    gemm_bf16<<<dim3(N / NBLK, M / MBLK, batch), dim3(256), 0, stream>>>(
        A, Bt, bias, res, pos, pD, pM, Cf, Cb, M, N, K, relu, sA, sB, sC);
  };

  // weights -> transposed bf16
  cvtT(gcn_W, wGCN, NF, D, 1);
  cvtT(fc1_W, wFC1, D, D, 1);
  for (int i = 0; i < 8; i++) cvtT(W[i], wAT[i], D, D, 1);
  cvtT(ff_W1, wFF1, D, DFF, 1);
  cvtT(ff_W2, wFF2, DFF, D, 1);
  // dec_in (B, NS, T*NF) -> per-batch transposed (B, T*NF, NS)
  cvtT(dec_in, decTb, NS, T * NF, B);

  // GCN, reassociated: agg = adj @ dec_in (per batch), then @ gcn_W
  adj_kernel<<<dim3(B), dim3(128), 0, stream>>>(locs, adjb);
  gemm(adjb, decTb, nullptr, nullptr, nullptr, 1, 1, nullptr, aggb,
       NS, T * NF, NS, 0, (long long)NS * NS, (long long)NS * T * NF, (long long)NS * T * NF, B);
  gemm(aggb, wGCN, gcn_b, nullptr, nullptr, 1, 1, nullptr, qb,
       (int)RT, D, NF, 1, 0, 0, 0, 1);
  gemm(qb, wFC1, fc1_b, nullptr, dec_pos, NS * T, T, x32, xb,
       (int)RT, D, D, 0, 0, 0, 0, 1);

  // self-attention
  gemm(xb, wAT[0], bb[0], nullptr, nullptr, 1, 1, nullptr, qb, (int)RT, D, D, 0, 0, 0, 0, 1);
  gemm(xb, wAT[1], bb[1], nullptr, nullptr, 1, 1, nullptr, kb, (int)RT, D, D, 0, 0, 0, 0, 1);
  gemm(xb, wAT[2], bb[2], nullptr, nullptr, 1, 1, nullptr, vb, (int)RT, D, D, 0, 0, 0, 0, 1);
  attn_kernel<<<dim3(B * NS, 8), dim3(32), 0, stream>>>(qb, kb, vb, mb, 1);
  gemm(mb, wAT[3], bb[3], x32, nullptr, 1, 1, y32, nullptr, (int)RT, D, D, 0, 0, 0, 0, 1);
  ln_kernel<<<dim3((unsigned)(RT / 8)), dim3(256), 0, stream>>>(y32, ln_g[0], ln_b[0], x32, xb);

  // cross-attention (mem = enc_out + enc_pos)
  addpos_bf16<<<dim3((unsigned)(XE / 256)), dim3(256), 0, stream>>>(
      enc_out, enc_pos, mb, XE, D, NS * T, T);
  gemm(xb, wAT[4], bb[4], nullptr, nullptr, 1, 1, nullptr, qb, (int)RT, D, D, 0, 0, 0, 0, 1);
  gemm(mb, wAT[5], bb[5], nullptr, nullptr, 1, 1, nullptr, kb, (int)RT, D, D, 0, 0, 0, 0, 1);
  gemm(mb, wAT[6], bb[6], nullptr, nullptr, 1, 1, nullptr, vb, (int)RT, D, D, 0, 0, 0, 0, 1);
  attn_kernel<<<dim3(B * NS, 8), dim3(32), 0, stream>>>(qb, kb, vb, mb, 0);
  gemm(mb, wAT[7], bb[7], x32, nullptr, 1, 1, y32, nullptr, (int)RT, D, D, 0, 0, 0, 0, 1);
  ln_kernel<<<dim3((unsigned)(RT / 8)), dim3(256), 0, stream>>>(y32, ln_g[1], ln_b[1], x32, xb);

  // feed-forward
  gemm(xb, wFF1, ff_b1, nullptr, nullptr, 1, 1, nullptr, h1, (int)RT, DFF, D, 1, 0, 0, 0, 1);
  gemm(h1, wFF2, ff_b2, x32, nullptr, 1, 1, y32, nullptr, (int)RT, D, DFF, 0, 0, 0, 0, 1);
  ln_kernel<<<dim3((unsigned)(RT / 8)), dim3(256), 0, stream>>>(y32, ln_g[2], ln_b[2], (float*)d_out, nullptr);
}
